// SoftmaxAttention_3805341024305
// MI455X (gfx1250) — compile-verified, matrix operand order checked by reading, not run on hardware
//
#include <hip/hip_runtime.h>

// ---------------------------------------------------------------------------
// Flash-style masked cross-attention for CDNA5 (gfx1250), bf16 WMMA pipeline.
//   out = masked_softmax(X·Yᵀ, ymask) · Y, scaled by xmask, per batch.
// Launched twice with (pre,hypo) and (hypo,pre) to produce both outputs.
// ---------------------------------------------------------------------------

typedef __attribute__((ext_vector_type(16))) __bf16    bf16x16;
typedef __attribute__((ext_vector_type(8)))  float     f32x8;
typedef __attribute__((ext_vector_type(4)))  float     f32x4;
typedef __attribute__((ext_vector_type(4)))  unsigned  u32x4;

union FragB {                 // one WMMA 16-bit A/B operand (16 bf16 per lane)
  bf16x16        v;
  u32x4          q[2];
  unsigned short s[16];
};

__device__ __forceinline__ unsigned short f2bf(float f) {
  union { float f; unsigned u; } x; x.f = f;
  unsigned r = x.u + 0x7FFFu + ((x.u >> 16) & 1u);   // round-to-nearest-even
  return (unsigned short)(r >> 16);
}

#define HDIM 256      // feature dim
#define NQ   32       // streamed N columns per step (= one bf16 WMMA K)
#define MW   16       // rows per wave (one WMMA M tile)
#define NW   4        // waves per block
#define MB   (MW*NW)  // rows per block

__launch_bounds__(NW * 32)
__global__ void attn_pass(const float* __restrict__ X,     // (B, Mtot, H)
                          const int*   __restrict__ xmask, // (B, Mtot)
                          const float* __restrict__ Y,     // (B, Ntot, H)
                          const int*   __restrict__ ymask, // (B, Ntot)
                          float*       __restrict__ Out,   // (B, Mtot, H)
                          int Mtot, int Ntot)
{
  __shared__ __align__(16) unsigned short sY [NQ * HDIM];   // (q, h) bf16, 16 KB
  __shared__ __align__(16) unsigned short sYT[HDIM * NQ];   // (h, q) bf16, 16 KB
  __shared__ __align__(16) unsigned short sP [NW * MW * NQ];// probs,      4 KB

  const int tid  = threadIdx.x;
  const int wave = tid >> 5;
  const int lane = tid & 31;
  const int ln   = lane & 15;     // N / M index within a tile half
  const int hb   = lane >> 4;     // which 16-lane half

  const int mtiles = Mtot / MB;
  const int b  = blockIdx.x / mtiles;
  const int m0 = (blockIdx.x % mtiles) * MB + wave * MW;

  const float* Xrow = X + ((size_t)b * Mtot + m0 + ln) * HDIM; // this lane's A row
  const float* Yb   = Y + (size_t)b * Ntot * HDIM;
  const int*   ymb  = ymask + (size_t)b * Ntot;

  f32x8 acc[16] = {};             // 16x256 fp32 output accumulator (128 VGPR)
  float rmax[8], rsum[8];
  #pragma unroll
  for (int r = 0; r < 8; ++r) { rmax[r] = 0.0f; rsum[r] = 0.0f; }

  for (int n0 = 0; n0 < Ntot; n0 += NQ) {
    // ---- stage Y chunk (32 x 256 fp32) -> LDS bf16, row-major + transposed
    #pragma unroll
    for (int j = 0; j < (NQ * HDIM / 4) / (NW * 32); ++j) {   // 16 float4/thread
      const int idx4 = j * (NW * 32) + tid;
      const int q    = idx4 >> 6;              // 64 float4 per row
      const int hc   = (idx4 & 63) << 2;
      const f32x4 f  = *(const f32x4*)(Yb + (size_t)(n0 + q) * HDIM + hc);
      const unsigned short b0 = f2bf(f.x), b1 = f2bf(f.y),
                           b2 = f2bf(f.z), b3 = f2bf(f.w);
      unsigned* dst = (unsigned*)&sY[q * HDIM + hc];
      dst[0] = (unsigned)b0 | ((unsigned)b1 << 16);
      dst[1] = (unsigned)b2 | ((unsigned)b3 << 16);
      sYT[(hc + 0) * NQ + q] = b0;
      sYT[(hc + 1) * NQ + q] = b1;
      sYT[(hc + 2) * NQ + q] = b2;
      sYT[(hc + 3) * NQ + q] = b3;
    }
    __syncthreads();

    if (n0 + NQ < Ntot)   // warm L2/GL1 for the next chunk
      __builtin_prefetch(Yb + (size_t)(n0 + NQ) * HDIM + tid * 16, 0, 1);

    // ---- S = X · Yᵀ : two 16x16 fp32 tiles over K = HDIM ----
    f32x8 s0 = {}, s1 = {};
    #pragma unroll
    for (int k0 = 0; k0 < HDIM; k0 += 32) {
      FragB a, bf0, bf1;
      const f32x4* ap = (const f32x4*)(Xrow + k0 + hb * 8);
      const f32x4 a0 = ap[0], a1 = ap[1], a2 = ap[4], a3 = ap[5];
      a.s[ 0]=f2bf(a0.x); a.s[ 1]=f2bf(a0.y); a.s[ 2]=f2bf(a0.z); a.s[ 3]=f2bf(a0.w);
      a.s[ 4]=f2bf(a1.x); a.s[ 5]=f2bf(a1.y); a.s[ 6]=f2bf(a1.z); a.s[ 7]=f2bf(a1.w);
      a.s[ 8]=f2bf(a2.x); a.s[ 9]=f2bf(a2.y); a.s[10]=f2bf(a2.z); a.s[11]=f2bf(a2.w);
      a.s[12]=f2bf(a3.x); a.s[13]=f2bf(a3.y); a.s[14]=f2bf(a3.z); a.s[15]=f2bf(a3.w);

      const u32x4* b0p = (const u32x4*)&sY[(ln)      * HDIM + k0 + hb * 8];
      const u32x4* b1p = (const u32x4*)&sY[(16 + ln) * HDIM + k0 + hb * 8];
      bf0.q[0] = b0p[0]; bf0.q[1] = b0p[2];
      bf1.q[0] = b1p[0]; bf1.q[1] = b1p[2];

      s0 = __builtin_amdgcn_wmma_f32_16x16x32_bf16(false, a.v, false, bf0.v,
                                                   (short)0, s0, false, false);
      s1 = __builtin_amdgcn_wmma_f32_16x16x32_bf16(false, a.v, false, bf1.v,
                                                   (short)0, s1, false, false);
    }

    // ---- masked online softmax (matches softmax(s*m)*m / (sum+EPS)) ----
    const float mv0 = (float)ymb[n0 + ln];
    const float mv1 = (float)ymb[n0 + 16 + ln];
    #pragma unroll
    for (int r = 0; r < 8; ++r) {
      float v0 = s0[r] * mv0;
      float v1 = s1[r] * mv1;
      float cm = fmaxf(v0, v1);
      #pragma unroll
      for (int d = 1; d < 16; d <<= 1)
        cm = fmaxf(cm, __shfl_xor(cm, d, 32));       // row max within half-wave
      const float nm = fmaxf(rmax[r], cm);
      const float sc = __expf(rmax[r] - nm);
      rmax[r] = nm;
      const float w0 = mv0 * __expf(v0 - nm);
      const float w1 = mv1 * __expf(v1 - nm);
      float rs = w0 + w1;
      #pragma unroll
      for (int d = 1; d < 16; d <<= 1)
        rs += __shfl_xor(rs, d, 32);                 // row sum within half-wave
      rsum[r] = rsum[r] * sc + rs;
      #pragma unroll
      for (int t = 0; t < 16; ++t) acc[t][r] *= sc;  // rescale running output
      // stash probabilities (C-layout -> row-major bf16) for A-frag reload
      const int pm = wave * (MW * NQ) + (r + 8 * hb) * NQ;
      sP[pm + ln]      = f2bf(w0);
      sP[pm + 16 + ln] = f2bf(w1);
    }

    // ---- acc += P(16x32) · V(32x16) per 16-wide H tile ----
    FragB p;
    const u32x4* pp = (const u32x4*)&sP[wave * (MW * NQ) + ln * NQ + hb * 8];
    p.q[0] = pp[0]; p.q[1] = pp[2];
    #pragma unroll
    for (int ht = 0; ht < 16; ++ht) {
      FragB vb;
      const u32x4* vp = (const u32x4*)&sYT[(ht * 16 + ln) * NQ + hb * 8];
      vb.q[0] = vp[0]; vb.q[1] = vp[2];
      acc[ht] = __builtin_amdgcn_wmma_f32_16x16x32_bf16(false, p.v, false, vb.v,
                                                        (short)0, acc[ht],
                                                        false, false);
    }
    __syncthreads();   // sY/sYT free for restaging
  }

  // ---- normalize, apply xmask, write fp32 output ----
  #pragma unroll
  for (int r = 0; r < 8; ++r) {
    const int   m   = m0 + r + 8 * hb;
    const float xm  = (float)xmask[(size_t)b * Mtot + m];
    const float inv = xm / (rsum[r] + 1e-13f);
    float* orow = Out + ((size_t)b * Mtot + m) * HDIM;
    #pragma unroll
    for (int ht = 0; ht < 16; ++ht)
      orow[ht * 16 + ln] = acc[ht][r] * inv;
  }
}

extern "C" void kernel_launch(void* const* d_in, const int* in_sizes, int n_in,
                              void* d_out, int out_size, void* d_ws, size_t ws_size,
                              hipStream_t stream) {
  const float* pre  = (const float*)d_in[0];
  const int*   pm   = (const int*)  d_in[1];
  const float* hypo = (const float*)d_in[2];
  const int*   hm   = (const int*)  d_in[3];
  float* out = (float*)d_out;

  const int B = 64, P = 1024, Q = 1024;
  dim3 blk(NW * 32);
  dim3 grid(B * (P / MB));
  // attented_pre = softmax_rows(pre·hypoᵀ, hm) · hypo  * pm
  attn_pass<<<grid, blk, 0, stream>>>(pre, pm, hypo, hm, out, P, Q);
  // attented_hypo = softmax_rows(hypo·preᵀ, pm) · pre  * hm  (perfect symmetry)
  attn_pass<<<grid, blk, 0, stream>>>(hypo, hm, pre, pm,
                                      out + (size_t)B * P * HDIM, Q, P);
}
